// NeuralTuringMachine_14877766713534
// MI455X (gfx1250) — compile-verified, matrix operand order checked by reading, not run on hardware
//
#include <hip/hip_runtime.h>
#include <hip/hip_bf16.h>
#include <cstdint>
#include <cstddef>

// Problem sizes
#define Bn 32
#define Sn 256
#define In 256
#define Cn 512
#define Nn 1024
#define Vn 64
#define On 256
#define G4n 2048            // 4*C
#define HOUTn 272           // 263 head outputs padded to 17 tiles of 16
#define HVn 576             // C + V
#define NWG 32
#define TPB 256             // 8 waves of 32
#define MST 66              // LDS memory row stride (floats), padded vs 64
#define EPSf 1e-12f

typedef __attribute__((ext_vector_type(16))) _Float16 v16h;
typedef __attribute__((ext_vector_type(8)))  _Float16 v8h;
typedef __attribute__((ext_vector_type(8)))  float    v8f;

// ---------------- workspace layout (bytes, all 32B-aligned) ----------------
#define WS_BAR    ((size_t)0)                        // 2 uints (cnt, gen)
#define WS_C      ((size_t)256)                      // 32*512 f32 (cell state)
#define WS_GATES  (WS_C     + (size_t)65536)         // 32*2048 f32
#define WS_HEADO  (WS_GATES + (size_t)262144)        // 32*272 f32
#define WS_BGATE  (WS_HEADO + (size_t)34816)         // 2048 f32
#define WS_BHEAD  (WS_BGATE + (size_t)8192)          // 272 f32 (pad to 1280B)
#define WS_BOUT   (WS_BHEAD + (size_t)1280)          // 256 f32
#define WS_H16    (WS_BOUT  + (size_t)1024)          // 32*512 f16
#define WS_RV16   (WS_H16   + (size_t)32768)         // 32*64 f16
#define WS_X16    (WS_RV16  + (size_t)4096)          // 32*256*256 f16
#define WS_WIH    (WS_X16   + (size_t)4194304)       // 320*2048 f16 (frag-major, KS=10)
#define WS_WHH    (WS_WIH   + (size_t)1310720)       // 512*2048 f16 (frag-major, KS=16)
#define WS_WHEAD  (WS_WHH   + (size_t)2097152)       // 512*272 f16 (frag-major, KS=16)
#define WS_WOUT   (WS_WHEAD + (size_t)278528)        // 576*256 f16 (frag-major, KS=18)
#define WS_END    (WS_WOUT  + (size_t)294912)

// LDS: 1024*66 + 1024 + 1024 + 1024 + 256 + 256 + 64 + 64 + 64 + 16 floats
#define SMEM_BYTES 285504

// ---------------- small math helpers ----------------
__device__ __forceinline__ float sigmoidf_(float x) { return 1.0f / (1.0f + expf(-x)); }
__device__ __forceinline__ float softplusf_(float x) { return (x > 20.0f) ? x : log1pf(expf(x)); }

// ---------------- WMMA tile helper ----------------
// One 16x16 fp32 output tile accumulated over K, all-f16 operands, f32 accum.
// A:  f16 row-major activations, lda in halves. Per CDNA5 16-bit A layout, lane L
//     (row M = L&15) needs K = k0 + ((e&8)<<1) + ((L&16)?8:0) + (e&7)  -> two
//     aligned 16B vector loads per k-step.
// Wf: f16 weights pre-packed FRAGMENT-MAJOR: for each (ntile,kstep) a 1KB block
//     of 32 lanes x 16 halves in exact fragment order -> one aligned 32B load.
__device__ __forceinline__ v8f wmma_acc(const _Float16* __restrict__ A, int lda,
                                        const _Float16* __restrict__ Wf,
                                        int ksteps, v8f acc) {
  const int lane = threadIdx.x & 31;
  const int rc   = lane & 15;
  const int kHi  = (lane & 16) ? 8 : 0;
  const _Float16* wp = Wf + lane * 16;
  const _Float16* ap = A + rc * lda + kHi;
  for (int ks = 0; ks < ksteps; ++ks) {
    if (ks + 1 < ksteps) __builtin_prefetch(wp + (size_t)(ks + 1) * 512, 0, 1);
    const v16h b   = *(const v16h*)(wp + (size_t)ks * 512);
    const v8h  alo = *(const v8h*)(ap + ks * 32);
    const v8h  ahi = *(const v8h*)(ap + ks * 32 + 16);
    const v16h a   = __builtin_shufflevector(alo, ahi, 0, 1, 2, 3, 4, 5, 6, 7,
                                             8, 9, 10, 11, 12, 13, 14, 15);
    acc = __builtin_amdgcn_wmma_f32_16x16x32_f16(false, a, false, b, (short)0, acc,
                                                 false, false);
  }
  return acc;
}

// Store 16x16 f32 D tile (+ per-column bias). out pre-offset to tile origin.
__device__ __forceinline__ void store_tile(float* __restrict__ out, int ldo,
                                           const float* __restrict__ bias, v8f acc) {
  const int lane  = threadIdx.x & 31;
  const int col   = lane & 15;
  const int rbase = (lane & 16) ? 8 : 0;
#pragma unroll
  for (int r = 0; r < 8; ++r)
    out[(rbase + r) * ldo + col] = acc[r] + bias[col];
}

// ---------------- grid barrier (persistent cooperative) ----------------
__device__ __forceinline__ void grid_barrier(unsigned* cnt, unsigned* gen) {
  __syncthreads();
  if (threadIdx.x == 0) {
    __threadfence();
    unsigned g = atomicAdd(gen, 0u);               // true L2 read
    if (atomicAdd(cnt, 1u) == NWG - 1) {
      atomicExch(cnt, 0u);
      __threadfence();
      atomicAdd(gen, 1u);
    } else {
      while (atomicAdd(gen, 0u) == g) { __builtin_amdgcn_s_sleep(1); }
    }
    __threadfence();
  }
  __syncthreads();
}

// ---------------- block reductions (256 threads) ----------------
__device__ __forceinline__ float block_sum(float v, float* red) {
  const int tid = threadIdx.x;
  red[tid] = v; __syncthreads();
  for (int s = 128; s >= 1; s >>= 1) {
    if (tid < s) red[tid] += red[tid + s];
    __syncthreads();
  }
  float r = red[0]; __syncthreads();
  return r;
}
__device__ __forceinline__ float block_max(float v, float* red) {
  const int tid = threadIdx.x;
  red[tid] = v; __syncthreads();
  for (int s = 128; s >= 1; s >>= 1) {
    if (tid < s) red[tid] = fmaxf(red[tid], red[tid + s]);
    __syncthreads();
  }
  float r = red[0]; __syncthreads();
  return r;
}

// ---------------- pack / init kernel ----------------
__device__ __forceinline__ float head_w(int n, int k,
    const float* rkW, const float* rsW, const float* rgW, const float* rshW,
    const float* rgaW, const float* wkW, const float* wsW, const float* weW,
    const float* waW) {
  if (n < 64)  return rkW[n * Cn + k];
  if (n < 128) return wkW[(n - 64) * Cn + k];
  if (n < 192) return weW[(n - 128) * Cn + k];
  if (n < 256) return waW[(n - 192) * Cn + k];
  if (n == 256) return rsW[k];
  if (n == 257) return rgW[k];
  if (n == 258) return rgaW[k];
  if (n == 259) return wsW[k];
  if (n < 263) return rshW[(n - 260) * Cn + k];
  return 0.0f;
}
__device__ __forceinline__ float head_b(int n,
    const float* rkb, const float* rsb, const float* rgb, const float* rshb,
    const float* rgab, const float* wkb, const float* wsb, const float* web,
    const float* wab) {
  if (n < 64)  return rkb[n];
  if (n < 128) return wkb[n - 64];
  if (n < 192) return web[n - 128];
  if (n < 256) return wab[n - 192];
  if (n == 256) return rsb[0];
  if (n == 257) return rgb[0];
  if (n == 258) return rgab[0];
  if (n == 259) return wsb[0];
  if (n < 263) return rshb[n - 260];
  return 0.0f;
}

// Decode fragment-major packed index -> (n, kk) source coordinates.
__device__ __forceinline__ void frag_decode(size_t i, int KS, int& n, int& kk) {
  const int e    = (int)(i & 15);
  const int lane = (int)((i >> 4) & 31);
  const int blk  = (int)(i >> 9);
  const int ks   = blk % KS;
  const int nt   = blk / KS;
  n  = nt * 16 + (lane & 15);
  kk = ks * 32 + ((e & 8) << 1) + ((lane & 16) ? 8 : 0) + (e & 7);
}

__global__ void ntm_pack_init(
    const float* __restrict__ x,
    const float* __restrict__ Wih, const float* __restrict__ Whh,
    const float* __restrict__ bih, const float* __restrict__ bhh,
    const float* __restrict__ rkW, const float* __restrict__ rkb,
    const float* __restrict__ rsW, const float* __restrict__ rsb,
    const float* __restrict__ rgW, const float* __restrict__ rgb,
    const float* __restrict__ rshW, const float* __restrict__ rshb,
    const float* __restrict__ rgaW, const float* __restrict__ rgab,
    const float* __restrict__ wkW, const float* __restrict__ wkb,
    const float* __restrict__ wsW, const float* __restrict__ wsb,
    const float* __restrict__ weW, const float* __restrict__ web,
    const float* __restrict__ waW, const float* __restrict__ wab,
    const float* __restrict__ outW, const float* __restrict__ outb,
    _Float16* __restrict__ x16,
    _Float16* __restrict__ wih16, _Float16* __restrict__ whh16,
    _Float16* __restrict__ whead16, _Float16* __restrict__ wout16,
    float* __restrict__ bgate, float* __restrict__ bhead, float* __restrict__ bout,
    float* __restrict__ c, _Float16* __restrict__ h16, _Float16* __restrict__ rv16,
    unsigned* __restrict__ bar) {
  const size_t i = (size_t)blockIdx.x * blockDim.x + threadIdx.x;
  if (i < (size_t)Bn * Sn * In) x16[i] = (_Float16)x[i];           // 2,097,152
  if (i < (size_t)320 * G4n) {                                     // 655,360
    int n, kk; frag_decode(i, 10, n, kk);
    wih16[i] = (_Float16)Wih[n * 320 + kk];
  }
  if (i < (size_t)Cn * G4n) {                                      // 1,048,576
    int n, kk; frag_decode(i, 16, n, kk);
    whh16[i] = (_Float16)Whh[n * Cn + kk];
  }
  if (i < (size_t)Cn * HOUTn) {                                    // 139,264
    int n, kk; frag_decode(i, 16, n, kk);
    whead16[i] = (_Float16)head_w(n, kk, rkW, rsW, rgW, rshW, rgaW, wkW, wsW, weW, waW);
  }
  if (i < (size_t)HVn * On) {                                      // 147,456
    int n, kk; frag_decode(i, 18, n, kk);
    wout16[i] = (_Float16)outW[n * HVn + kk];
  }
  if (i < G4n)   bgate[i] = bih[i] + bhh[i];
  if (i < HOUTn) bhead[i] = head_b((int)i, rkb, rsb, rgb, rshb, rgab, wkb, wsb, web, wab);
  if (i < On)    bout[i] = outb[i];
  if (i < (size_t)Bn * Cn) { c[i] = 0.0f; h16[i] = (_Float16)0.0f; }
  if (i < (size_t)Bn * Vn) rv16[i] = (_Float16)0.0f;
  if (i < 2) bar[i] = 0u;
}

// ---------------- persistent NTM kernel ----------------
__global__ __launch_bounds__(TPB, 1) void ntm_persistent(
    const _Float16* __restrict__ x16,
    const _Float16* __restrict__ wih16, const _Float16* __restrict__ whh16,
    const _Float16* __restrict__ whead16, const _Float16* __restrict__ wout16,
    const float* __restrict__ bgate, const float* __restrict__ bhead,
    const float* __restrict__ bout,
    float* __restrict__ c, _Float16* __restrict__ h16, _Float16* __restrict__ rv16,
    float* __restrict__ gates, float* __restrict__ heado,
    unsigned* __restrict__ bar, float* __restrict__ dout) {
  extern __shared__ float smem[];
  float* MEM  = smem;                       // [1024][MST]
  float* INVN = MEM  + Nn * MST;            // 1024
  float* CWs  = INVN + Nn;                  // 1024 (sim/cw/ww scratch)
  float* FWs  = CWs  + Nn;                  // 1024 (fw)
  float* RED  = FWs  + Nn;                  // 256
  float* PART = RED  + TPB;                 // 256
  float* KEY  = PART + TPB;                 // 64
  float* ER   = KEY  + Vn;                  // 64
  float* AD   = ER   + Vn;                  // 64
  float* SC   = AD   + Vn;                  // 16 scalars

  const int tid  = threadIdx.x;
  const int bWG  = blockIdx.x;              // one workgroup per batch element
  const int wave = tid >> 5;
  const int slot = bWG * 8 + wave;          // 256 wave slots grid-wide
  unsigned* cnt = bar;
  unsigned* gen = bar + 1;

  // memory starts at zero (reference)
  for (int i = tid; i < Nn * MST; i += TPB) MEM[i] = 0.0f;
  __syncthreads();

  for (int t = 0; t < Sn; ++t) {
    // ---- P1: gate GEMM  gates[32,2048] = [x_t | rv] @ WihT + h @ WhhT + b
    {
      const int tile = slot;                // 256 tiles == 256 slots
      const int mt = tile & 1, nt = tile >> 1;
      v8f acc = {};
      acc = wmma_acc(x16 + (size_t)(mt * 16) * (Sn * In) + (size_t)t * In, Sn * In,
                     wih16 + (size_t)(nt * 10) * 512, 8, acc);
      acc = wmma_acc(rv16 + mt * 16 * Vn, Vn,
                     wih16 + (size_t)(nt * 10 + 8) * 512, 2, acc);
      acc = wmma_acc(h16 + mt * 16 * Cn, Cn,
                     whh16 + (size_t)(nt * 16) * 512, 16, acc);
      store_tile(gates + (mt * 16) * G4n + nt * 16, G4n, bgate + nt * 16, acc);
    }
    grid_barrier(cnt, gen);

    // ---- P2: LSTM cell elementwise (writes f16 h for all GEMM consumers)
    for (int e = bWG * TPB + tid; e < Bn * Cn; e += NWG * TPB) {
      const int bb = e >> 9, j = e & 511;
      const float* g = gates + bb * G4n;
      const float ig = sigmoidf_(g[j]);
      const float fg = sigmoidf_(g[Cn + j]);
      const float gg = tanhf(g[2 * Cn + j]);
      const float og = sigmoidf_(g[3 * Cn + j]);
      const float cc = fg * c[e] + ig * gg;
      c[e] = cc;
      h16[e] = (_Float16)(og * tanhf(cc));
    }
    grid_barrier(cnt, gen);

    // ---- P3: head GEMM  heado[32,272] = h @ WheadT + b
    for (int tile = slot; tile < 34; tile += NWG * 8) {
      const int mt = tile & 1, nt = tile >> 1;
      v8f acc = {};
      acc = wmma_acc(h16 + mt * 16 * Cn, Cn, whead16 + (size_t)(nt * 16) * 512, 16, acc);
      store_tile(heado + (mt * 16) * HOUTn + nt * 16, HOUTn, bhead + nt * 16, acc);
    }
    grid_barrier(cnt, gen);

    // ---- P4: per-batch memory phase (LDS resident)
    {
      const float* ho = heado + bWG * HOUTn;
      if (tid < Vn) {
        KEY[tid] = tanhf(ho[tid]);                 // read key
        ER[tid]  = sigmoidf_(ho[128 + tid]);       // erase
        AD[tid]  = tanhf(ho[192 + tid]);           // add
      }
      if (tid == 64) SC[1] = softplusf_(ho[256]);          // read strength
      if (tid == 65) SC[2] = sigmoidf_(ho[257]);           // gate
      if (tid == 66) SC[3] = 1.0f + softplusf_(ho[258]);   // gamma
      if (tid == 67) SC[4] = softplusf_(ho[259]);          // write strength
      __syncthreads();
      if (tid == 0) {
        float s = 0.0f;
        for (int v = 0; v < Vn; ++v) s += KEY[v] * KEY[v];
        SC[0] = 1.0f / fmaxf(sqrtf(s), EPSf);
      }
      __syncthreads();

      // read-head content addressing (also caches row inverse norms)
      float lmax = -3.4e38f;
      {
        const float invk = SC[0], str = SC[1];
        for (int i = 0; i < 4; ++i) {
          const int n = tid + i * TPB;
          const float* mr = MEM + n * MST;
          float d = 0.0f, q = 0.0f;
          for (int v = 0; v < Vn; ++v) { const float m = mr[v]; d += m * KEY[v]; q += m * m; }
          const float invn = 1.0f / fmaxf(sqrtf(q), EPSf);
          INVN[n] = invn;
          const float l = str * d * invn * invk;
          CWs[n] = l;
          lmax = fmaxf(lmax, l);
        }
      }
      const float mx = block_max(lmax, RED);
      float lsum = 0.0f;
      for (int i = 0; i < 4; ++i) {
        const int n = tid + i * TPB;
        const float e = expf(CWs[n] - mx);
        CWs[n] = e; lsum += e;
      }
      const float invS = 1.0f / block_sum(lsum, RED);
      float psum = 0.0f;
      {
        const float gamma = SC[3];
        for (int i = 0; i < 4; ++i) {
          const int n = tid + i * TPB;
          const float cw = CWs[n] * invS;
          CWs[n] = cw;
          const float p = powf(cw, gamma);
          FWs[n] = p; psum += p;
        }
      }
      const float invP = 1.0f / block_sum(psum, RED);
      {
        const float gate = SC[2];
        for (int i = 0; i < 4; ++i) {
          const int n = tid + i * TPB;
          FWs[n] = gate * FWs[n] * invP + (1.0f - gate) * CWs[n];
        }
      }
      __syncthreads();

      // read vector: rv[b] = fw @ MEM
      {
        const int v = tid & 63, ch = tid >> 6;
        float acc = 0.0f;
        for (int n = ch * 256; n < ch * 256 + 256; ++n) acc += FWs[n] * MEM[n * MST + v];
        PART[tid] = acc;
      }
      __syncthreads();
      if (tid < Vn)
        rv16[bWG * Vn + tid] =
            (_Float16)(PART[tid] + PART[64 + tid] + PART[128 + tid] + PART[192 + tid]);
      __syncthreads();

      // write-head content addressing (memory unchanged; reuse INVN)
      if (tid < Vn) KEY[tid] = tanhf(ho[64 + tid]);
      __syncthreads();
      if (tid == 0) {
        float s = 0.0f;
        for (int v = 0; v < Vn; ++v) s += KEY[v] * KEY[v];
        SC[0] = 1.0f / fmaxf(sqrtf(s), EPSf);
      }
      __syncthreads();
      float lmax2 = -3.4e38f;
      {
        const float invk = SC[0], str = SC[4];
        for (int i = 0; i < 4; ++i) {
          const int n = tid + i * TPB;
          const float* mr = MEM + n * MST;
          float d = 0.0f;
          for (int v = 0; v < Vn; ++v) d += mr[v] * KEY[v];
          const float l = str * d * INVN[n] * invk;
          CWs[n] = l;
          lmax2 = fmaxf(lmax2, l);
        }
      }
      const float mx2 = block_max(lmax2, RED);
      float lsum2 = 0.0f;
      for (int i = 0; i < 4; ++i) {
        const int n = tid + i * TPB;
        const float e = expf(CWs[n] - mx2);
        CWs[n] = e; lsum2 += e;
      }
      const float invS2 = 1.0f / block_sum(lsum2, RED);
      for (int i = 0; i < 4; ++i) CWs[tid + i * TPB] *= invS2;   // ww
      __syncthreads();

      // memory update: M = M*(1 - ww⊗erase) + ww⊗add
      for (int i = 0; i < 256; ++i) {
        const int e = tid + i * TPB;
        const int n = e >> 6, v = e & 63;
        const float w = CWs[n];
        float* p = MEM + n * MST + v;
        *p = *p * (1.0f - w * ER[v]) + w * AD[v];
      }
      __syncthreads();
    }
    grid_barrier(cnt, gen);

    // ---- P5: output GEMM  out[32,256] = [h | rv] @ WoutT + b  -> dout[b, t, :]
    for (int tile = slot; tile < 32; tile += NWG * 8) {
      const int mt = tile & 1, nt = tile >> 1;
      v8f acc = {};
      acc = wmma_acc(h16 + mt * 16 * Cn, Cn, wout16 + (size_t)(nt * 18) * 512, 16, acc);
      acc = wmma_acc(rv16 + mt * 16 * Vn, Vn, wout16 + (size_t)(nt * 18 + 16) * 512, 2, acc);
      store_tile(dout + (size_t)(mt * 16) * (Sn * On) + (size_t)t * On + nt * 16,
                 Sn * On, bout + nt * 16, acc);
    }
    grid_barrier(cnt, gen);
  }

  // final memory dump: dout[B*S*O + b*N*V + ...]
  {
    float* md = dout + (size_t)Bn * Sn * On + (size_t)bWG * Nn * Vn;
    for (int i = 0; i < 256; ++i) {
      const int e = tid + i * TPB;
      md[e] = MEM[(e >> 6) * MST + (e & 63)];
    }
  }
}

// ---------------- host launcher ----------------
extern "C" void kernel_launch(void* const* d_in, const int* in_sizes, int n_in,
                              void* d_out, int out_size, void* d_ws, size_t ws_size,
                              hipStream_t stream) {
  (void)in_sizes; (void)n_in; (void)out_size; (void)ws_size;
  const float* x    = (const float*)d_in[0];
  const float* Wih  = (const float*)d_in[1];
  const float* Whh  = (const float*)d_in[2];
  const float* bih  = (const float*)d_in[3];
  const float* bhh  = (const float*)d_in[4];
  const float* rkW  = (const float*)d_in[5];
  const float* rkb  = (const float*)d_in[6];
  const float* rsW  = (const float*)d_in[7];
  const float* rsb  = (const float*)d_in[8];
  const float* rgW  = (const float*)d_in[9];
  const float* rgb  = (const float*)d_in[10];
  const float* rshW = (const float*)d_in[11];
  const float* rshb = (const float*)d_in[12];
  const float* rgaW = (const float*)d_in[13];
  const float* rgab = (const float*)d_in[14];
  const float* wkW  = (const float*)d_in[15];
  const float* wkb  = (const float*)d_in[16];
  const float* wsW  = (const float*)d_in[17];
  const float* wsb  = (const float*)d_in[18];
  const float* weW  = (const float*)d_in[19];
  const float* web  = (const float*)d_in[20];
  const float* waW  = (const float*)d_in[21];
  const float* wab  = (const float*)d_in[22];
  const float* outW = (const float*)d_in[23];
  const float* outb = (const float*)d_in[24];

  char* ws = (char*)d_ws;
  unsigned* bar     = (unsigned*)(ws + WS_BAR);
  float*    c       = (float*)(ws + WS_C);
  float*    gates   = (float*)(ws + WS_GATES);
  float*    heado   = (float*)(ws + WS_HEADO);
  float*    bgate   = (float*)(ws + WS_BGATE);
  float*    bhead   = (float*)(ws + WS_BHEAD);
  float*    bout    = (float*)(ws + WS_BOUT);
  _Float16* h16     = (_Float16*)(ws + WS_H16);
  _Float16* rv16    = (_Float16*)(ws + WS_RV16);
  _Float16* x16     = (_Float16*)(ws + WS_X16);
  _Float16* wih16   = (_Float16*)(ws + WS_WIH);
  _Float16* whh16   = (_Float16*)(ws + WS_WHH);
  _Float16* whead16 = (_Float16*)(ws + WS_WHEAD);
  _Float16* wout16  = (_Float16*)(ws + WS_WOUT);
  float* dout = (float*)d_out;

  (void)hipFuncSetAttribute((const void*)ntm_persistent,
                            hipFuncAttributeMaxDynamicSharedMemorySize, SMEM_BYTES);

  // pack weights to fragment-major f16, convert x to f16, fuse biases,
  // zero recurrent state + barrier (deterministic every call)
  ntm_pack_init<<<8192, 256, 0, stream>>>(
      x, Wih, Whh, bih, bhh, rkW, rkb, rsW, rsb, rgW, rgb, rshW, rshb, rgaW, rgab,
      wkW, wkb, wsW, wsb, weW, web, waW, wab, outW, outb,
      x16, wih16, whh16, whead16, wout16, bgate, bhead, bout, c, h16, rv16, bar);

  // persistent cooperative kernel: 32 WGs (1 per batch), 8 waves each,
  // 285KB LDS per WG holding the [N,V] memory matrix for all 256 steps
  ntm_persistent<<<NWG, TPB, SMEM_BYTES, stream>>>(
      x16, wih16, whh16, whead16, wout16, bgate, bhead, bout,
      c, h16, rv16, gates, heado, bar, dout);
}
